// _DenseGATLayer_47725676593310
// MI455X (gfx1250) — matrix operand order repaired and checked by reading, not compiled, not run on hardware
//
#include <hip/hip_runtime.h>
#include <hip/hip_bf16.h>

// Problem constants (from reference)
#define NN      4096   // nodes
#define IN_DIM  256
#define OUT_DIM 64
#define HEADS   8
#define HD      512    // HEADS*OUT_DIM

typedef __attribute__((ext_vector_type(16))) _Float16 v16h;
typedef __attribute__((ext_vector_type(8)))  float    v8f;
typedef __attribute__((ext_vector_type(2)))  float    v2f;

__device__ __forceinline__ float lrelu02(float s) {
    return s >= 0.0f ? s : 0.2f * s;
}

// Generic (__shared__) pointer -> 32-bit LDS byte offset (wave LDS base relative),
// as required by the VDST operand of global_load_async_to_lds_*.
__device__ __forceinline__ unsigned lds_off(const void* p) {
    return (unsigned)(unsigned long long)(__attribute__((address_space(3))) const void*)p;
}

// ---------------------------------------------------------------------------
// Kernel 1: h = x @ W_proj^T   (fp32 WMMA 16x16x4), also emit h^T in f16.
// Wave tile = 16x16 of h.  Grid: 256 Mtiles * 32 Ntiles = 8192 waves.
// A 16x4 f32 layout: lanes0-15 M=lane {V0:K0,V1:K1}; lanes16-31 {V0:K2,V1:K3}
// B 4x16 f32 layout (mirror): lanes0-15 N=lane {V0:K0,V1:K1}; lanes16-31 {K2,K3}
// C: VGPR v: lanes0-15 -> M=v, N=lane; lanes16-31 -> M=v+8, N=lane-16
// ---------------------------------------------------------------------------
__global__ __launch_bounds__(256) void gat_proj_kernel(
    const float* __restrict__ x, const float* __restrict__ W,
    float* __restrict__ h, _Float16* __restrict__ ht) {
    const int wave = threadIdx.x >> 5;
    const int lane = threadIdx.x & 31;
    const int hi   = lane >> 4;
    const int l15  = lane & 15;
    const int w    = blockIdx.x * 8 + wave;   // 0..8191
    const int i0   = (w >> 5) * 16;           // M tile
    const int n0   = (w & 31) * 16;           // N tile

    const float* xrow = x + (i0 + l15) * IN_DIM + 2 * hi;
    const float* wrow = W + (n0 + l15) * IN_DIM + 2 * hi;

    v8f c = {};
#pragma unroll 4
    for (int k = 0; k < IN_DIM; k += 4) {
        v2f a, b;
        a.x = xrow[k]; a.y = xrow[k + 1];
        b.x = wrow[k]; b.y = wrow[k + 1];
        c = __builtin_amdgcn_wmma_f32_16x16x4_f32(false, a, false, b,
                                                  (short)0, c, false, false);
    }
#pragma unroll
    for (int v = 0; v < 8; ++v) {
        const int row = i0 + v + 8 * hi;
        const int col = n0 + l15;
        h[row * HD + col]  = c[v];
        ht[col * NN + row] = (_Float16)c[v];
    }
}

// ---------------------------------------------------------------------------
// Kernel 2: e_src[h][n] = <h[n,h,:], attn_src[h,:]>, same for e_dst.
// Stored head-major for coalesced streaming in the attention kernel.
// ---------------------------------------------------------------------------
__global__ __launch_bounds__(256) void gat_edge_kernel(
    const float* __restrict__ h, const float* __restrict__ asrc,
    const float* __restrict__ adst, float* __restrict__ es,
    float* __restrict__ ed) {
    const int t  = blockIdx.x * 256 + threadIdx.x;   // 32768 (n, head) pairs
    const int n  = t >> 3;
    const int hh = t & 7;
    const float* hp = h + n * HD + hh * OUT_DIM;
    const float* sp = asrc + hh * OUT_DIM;
    const float* dp = adst + hh * OUT_DIM;
    float a = 0.0f, b = 0.0f;
#pragma unroll 8
    for (int d = 0; d < OUT_DIM; ++d) {
        const float v = hp[d];
        a += v * sp[d];
        b += v * dp[d];
    }
    es[hh * NN + n] = a;
    ed[hh * NN + n] = b;
}

// ---------------------------------------------------------------------------
// Kernel 3: maxdst[h] = max_j e_dst[h][j]   (upper bound for masked row max)
// ---------------------------------------------------------------------------
__global__ __launch_bounds__(256) void gat_maxdst_kernel(
    const float* __restrict__ ed, float* __restrict__ md) {
    __shared__ float red[256];
    const int hh = blockIdx.x;
    float m = -3.0e38f;
    for (int j = threadIdx.x; j < NN; j += 256)
        m = fmaxf(m, ed[hh * NN + j]);
    red[threadIdx.x] = m;
    __syncthreads();
    for (int s = 128; s > 0; s >>= 1) {
        if (threadIdx.x < s)
            red[threadIdx.x] = fmaxf(red[threadIdx.x], red[threadIdx.x + s]);
        __syncthreads();
    }
    if (threadIdx.x == 0) md[hh] = red[0];
}

// ---------------------------------------------------------------------------
// Kernel 4: fused masked-softmax attention + aggregation.
// Block = 256 threads = 8 waves (wave <-> head). Each wave owns two 16-row
// i-tiles (32 rows per block), loops over j in chunks of 32 using
// v_wmma_f32_16x16x32_f16:  A = p (f16 probabilities), B = h^T tile (f16).
// The 32x32 adjacency tile (the one unavoidable HBM stream) is double-
// buffered in LDS via global_load_async_to_lds_b128 (ASYNCcnt-tracked DMA,
// no VGPR staging) and synchronized with s_wait_asynccnt + barriers.
// A f16 16x32 layout: lane (hi,l15): M=l15, element e -> K=e+16*hi
// B f16 32x16 layout: lane (hi,l15): N=l15, element e -> K=e+16*hi
// ---------------------------------------------------------------------------
__global__ __launch_bounds__(256) void gat_attn_kernel(
    const int* __restrict__ adj, const _Float16* __restrict__ ht,
    const float* __restrict__ es, const float* __restrict__ ed,
    const float* __restrict__ md, const float* __restrict__ bias,
    float* __restrict__ out) {
    __shared__ int   adjLds[2][32][36];   // padded rows: 16B-aligned writes,
                                          // conflict-free column reads
    __shared__ float rsLds[HEADS][2][16];

    const int head = threadIdx.x >> 5;
    const int lane = threadIdx.x & 31;
    const int hi   = lane >> 4;
    const int l15  = lane & 15;
    const int i0   = blockIdx.x * 32;

    const float mdh = md[head];
    const float esA = es[head * NN + i0 + l15];
    const float esB = es[head * NN + i0 + 16 + l15];
    const float mA  = lrelu02(esA + mdh);
    const float mB  = lrelu02(esB + mdh);

    const float*    edh = ed + head * NN;
    const _Float16* htb = ht + (head * OUT_DIM + l15) * NN;

    // Per-thread slice of the adjacency tile DMA: 16 bytes each.
    const int rr = threadIdx.x >> 3;
    const int cc = (threadIdx.x & 7) * 4;
    const int* gsrc = adj + (i0 + rr) * NN + cc;

    v8f acc[8] = {};                      // [ntile][itile] -> acc[nt*2+it]
    float rsA = 0.0f, rsB = 0.0f;

    // Prologue: async-copy chunk 0 into buffer 0.
    asm volatile("global_load_async_to_lds_b128 %0, %1, off"
                 :: "v"(lds_off(&adjLds[0][rr][cc])),
                    "v"((unsigned long long)gsrc)
                 : "memory");

    const int NCH = NN / 32;
    for (int t = 0; t < NCH; ++t) {
        const int j0   = t * 32;
        const int cbuf = t & 1;

        if (t + 1 < NCH) {
            // Kick off next chunk's DMA into the other buffer, then wait for
            // the *current* chunk only (async loads complete in order).
            asm volatile("global_load_async_to_lds_b128 %0, %1, off"
                         :: "v"(lds_off(&adjLds[cbuf ^ 1][rr][cc])),
                            "v"((unsigned long long)(gsrc + j0 + 32))
                         : "memory");
            asm volatile("s_wait_asynccnt 0x1" ::: "memory");
        } else {
            asm volatile("s_wait_asynccnt 0x0" ::: "memory");
        }
        __syncthreads();   // current tile visible to all 8 head-waves

        // --- e_dst values for this chunk, ordered to match A's K layout ---
        float edv[16];
        *(float4*)&edv[0]  = *(const float4*)&edh[j0 + hi * 16];
        *(float4*)&edv[4]  = *(const float4*)&edh[j0 + hi * 16 + 4];
        *(float4*)&edv[8]  = *(const float4*)&edh[j0 + hi * 16 + 8];
        *(float4*)&edv[12] = *(const float4*)&edh[j0 + hi * 16 + 12];

        // --- build A (probabilities) for both i-tiles ---
        v16h aA, aB;
#pragma unroll
        for (int e = 0; e < 16; ++e) {
            const int k = e + (hi << 4);
            float sA = lrelu02(esA + edv[e]);
            float sB = lrelu02(esB + edv[e]);
            float pA = adjLds[cbuf][l15][k]      ? __expf(sA - mA) : 0.0f;
            float pB = adjLds[cbuf][16 + l15][k] ? __expf(sB - mB) : 0.0f;
            const _Float16 qA = (_Float16)pA;
            const _Float16 qB = (_Float16)pB;
            aA[e] = qA; aB[e] = qB;
            rsA += (float)qA;             // normalize with the f16-rounded p
            rsB += (float)qB;
        }

        // --- B tiles (h^T, contiguous 32B per lane) + 8 WMMAs per chunk ---
#pragma unroll
        for (int nt = 0; nt < 4; ++nt) {
            if (j0 + 32 < NN)   // prefetch next chunk's B rows into caches
                __builtin_prefetch(&htb[nt * 16 * NN + j0 + 32 + hi * 16], 0, 0);
            const v16h b =
                *(const v16h*)&htb[nt * 16 * NN + j0 + hi * 16];
            acc[nt * 2 + 0] = __builtin_amdgcn_wmma_f32_16x16x32_f16(
                false, aA, false, b, (short)0, acc[nt * 2 + 0], false, false);
            acc[nt * 2 + 1] = __builtin_amdgcn_wmma_f32_16x16x32_f16(
                false, aB, false, b, (short)0, acc[nt * 2 + 1], false, false);
        }
        __syncthreads();   // all waves done with this buffer before DMA reuse
    }

    // --- row-sum reduction (lanes L and L^16 cover all 32 K each chunk) ---
    rsA += __shfl_xor(rsA, 16, 32);
    rsB += __shfl_xor(rsB, 16, 32);
    if (hi == 0) {
        rsLds[head][0][l15] = rsA;
        rsLds[head][1][l15] = rsB;
    }
    __syncthreads();

    // --- normalize, add bias, store (C layout: VGPR v -> M = v + 8*hi) ---
#pragma unroll
    for (int nt = 0; nt < 4; ++nt) {
        const int d  = head * OUT_DIM + nt * 16 + l15;
        const float bv = bias[d];
#pragma unroll
        for (int v = 0; v < 8; ++v) {
            const int r = v + 8 * hi;
            const float ia = 1.0f / fmaxf(rsLds[head][0][r], 1e-30f);
            const float ib = 1.0f / fmaxf(rsLds[head][1][r], 1e-30f);
            out[(i0 + r) * HD + d]      = acc[nt * 2 + 0][v] * ia + bv;
            out[(i0 + 16 + r) * HD + d] = acc[nt * 2 + 1][v] * ib + bv;
        }
    }
}

// ---------------------------------------------------------------------------
extern "C" void kernel_launch(void* const* d_in, const int* in_sizes, int n_in,
                              void* d_out, int out_size, void* d_ws,
                              size_t ws_size, hipStream_t stream) {
    const float* x    = (const float*)d_in[0];   // [4096,256]
    const int*   adj  = (const int*)d_in[1];     // [4096,4096]
    const float* W    = (const float*)d_in[2];   // [512,256]
    const float* asrc = (const float*)d_in[3];   // [8,64]
    const float* adst = (const float*)d_in[4];   // [8,64]
    const float* bias = (const float*)d_in[5];   // [512]
    float* out = (float*)d_out;                  // [4096,512]

    char* ws = (char*)d_ws;
    float*    h  = (float*)ws;                                   // 8 MB
    _Float16* ht = (_Float16*)(ws + (size_t)8 * 1024 * 1024);    // 4 MB
    float*    es = (float*)(ws + (size_t)12 * 1024 * 1024);      // 128 KB
    float*    ed = (float*)(ws + (size_t)12 * 1024 * 1024 + 128 * 1024);
    float*    md = (float*)(ws + (size_t)12 * 1024 * 1024 + 256 * 1024);

    gat_proj_kernel<<<1024, 256, 0, stream>>>(x, W, h, ht);
    gat_edge_kernel<<<128, 256, 0, stream>>>(h, asrc, adst, es, ed);
    gat_maxdst_kernel<<<HEADS, 256, 0, stream>>>(ed, md);
    gat_attn_kernel<<<128, 256, 0, stream>>>(adj, ht, es, ed, md, bias, out);
}
